// Mask_Model_40037685134045
// MI455X (gfx1250) — matrix operand order repaired and checked by reading, not compiled, not run on hardware
//
#include <hip/hip_runtime.h>
#include <math.h>

// Problem sizes (fixed by the reference setup_inputs()).
static constexpr int  U_ROWS  = 50000;
static constexpr int  I_ROWS  = 25000;
static constexpr int  M_ROWS  = U_ROWS + I_ROWS;   // 75000 combined rows
static constexpr int  D_DIM   = 256;               // embed dim (GEMM K)
static constexpr int  H_DIM   = 128;               // head dim
static constexpr int  N_DIM   = 2 * H_DIM;         // 256 = [Q | K] columns
static constexpr long E_EDGES = 1600000;
static constexpr float INV_TAU = 5.0f;             // 1 / 0.2

typedef __attribute__((ext_vector_type(16))) __bf16 v16bf;
typedef __attribute__((ext_vector_type(8)))  float  v8f;

// ---------------------------------------------------------------------------
// float atomic max via int/uint ordering trick
// ---------------------------------------------------------------------------
__device__ __forceinline__ void atomicMaxFloat(float* addr, float val) {
  if (!(val < 0.0f)) {
    atomicMax((int*)addr, __float_as_int(val));
  } else {
    atomicMin((unsigned int*)addr, __float_as_uint(val));
  }
}

// ---------------------------------------------------------------------------
// Phase 0: build transposed bf16 weight [N=256][K=256] = [Wq | Wk]^T,
//          combined bias bc[256], and init row max / row sum accumulators.
// ---------------------------------------------------------------------------
__global__ void prep_kernel(const float* __restrict__ Wq, const float* __restrict__ bq,
                            const float* __restrict__ Wk, const float* __restrict__ bk,
                            __bf16* __restrict__ WcT, float* __restrict__ bc,
                            float* __restrict__ rmax_u, float* __restrict__ rsum_u,
                            float* __restrict__ rmax_i, float* __restrict__ rsum_i) {
  int idx = blockIdx.x * blockDim.x + threadIdx.x;
  if (idx < N_DIM * D_DIM) {
    int n = idx >> 8;          // output column 0..255
    int k = idx & 255;         // input dim
    float w = (n < H_DIM) ? Wq[k * H_DIM + n] : Wk[k * H_DIM + (n - H_DIM)];
    WcT[(size_t)n * D_DIM + k] = (__bf16)w;
  }
  if (idx < N_DIM) bc[idx] = (idx < H_DIM) ? bq[idx] : bk[idx - H_DIM];
  if (idx < U_ROWS) { rmax_u[idx] = -INFINITY; rsum_u[idx] = 0.0f; }
  if (idx < I_ROWS) { rmax_i[idx] = -INFINITY; rsum_i[idx] = 0.0f; }
}

// ---------------------------------------------------------------------------
// Phase 1: P[75000x256] = [user;item] @ [Wq|Wk] + [bq|bk]  via bf16 WMMA.
// One wave per 16x16 tile, 8 x v_wmma_f32_16x16x32_bf16 over K=256.
// ---------------------------------------------------------------------------
__global__ void gemm_qk_kernel(const float* __restrict__ user_embed,
                               const float* __restrict__ item_embed,
                               const __bf16* __restrict__ WcT,
                               const float* __restrict__ bc,
                               float* __restrict__ P) {
  const int wave  = threadIdx.x >> 5;
  const int lane  = threadIdx.x & 31;
  const int tile  = blockIdx.x * (blockDim.x >> 5) + wave;
  const int m0    = (tile >> 4) * 16;           // 16 N-tiles per M-tile row
  const int n0    = (tile & 15) * 16;
  if (m0 >= M_ROWS) return;                     // uniform per wave

  const int ml    = lane & 15;                  // A: M index within tile
  const int khalf = lane >> 4;                  // which K-half this lane holds
  const int nl    = lane & 15;                  // B/C/D: N index within tile

  int mr = m0 + ml;
  if (mr >= M_ROWS) mr = M_ROWS - 1;            // clamp for tail tile (stores guarded)
  const float*  arow = (mr < U_ROWS) ? (user_embed + (size_t)mr * D_DIM)
                                     : (item_embed + (size_t)(mr - U_ROWS) * D_DIM);
  const __bf16* brow = WcT + (size_t)(n0 + nl) * D_DIM;

  v8f c;
  const float bias = bc[n0 + nl];
#pragma unroll
  for (int j = 0; j < 8; ++j) c[j] = bias;

#pragma unroll
  for (int k0 = 0; k0 < D_DIM; k0 += 32) {
    // A fragment (16-bit A 16x32 layout): lane holds K = k0+khalf*16+{0..15}
    // (contiguous, matching B); convert f32 -> bf16 in-register.
    const float* ap = arow + k0 + khalf * 16;
    float4 av0 = *(const float4*)(ap);
    float4 av1 = *(const float4*)(ap + 4);
    float4 av2 = *(const float4*)(ap + 8);
    float4 av3 = *(const float4*)(ap + 12);
    v16bf a;
    a[0]  = (__bf16)av0.x; a[1]  = (__bf16)av0.y; a[2]  = (__bf16)av0.z; a[3]  = (__bf16)av0.w;
    a[4]  = (__bf16)av1.x; a[5]  = (__bf16)av1.y; a[6]  = (__bf16)av1.z; a[7]  = (__bf16)av1.w;
    a[8]  = (__bf16)av2.x; a[9]  = (__bf16)av2.y; a[10] = (__bf16)av2.z; a[11] = (__bf16)av2.w;
    a[12] = (__bf16)av3.x; a[13] = (__bf16)av3.y; a[14] = (__bf16)av3.z; a[15] = (__bf16)av3.w;

    // B fragment (16-bit B 32x16): lane nl holds column n0+nl,
    // K = k0 + khalf*16 + {0..15}; contiguous in the pre-transposed WcT.
    v16bf b = *(const v16bf*)(brow + k0 + khalf * 16);

    c = __builtin_amdgcn_wmma_f32_16x16x32_bf16(
            /*neg_a=*/false, a, /*neg_b=*/false, b,
            /*c_mod=*/(short)0, c, /*reuse_a=*/false, /*reuse_b=*/false);
  }

  // D layout: VGPR j -> M = m0 + 8*khalf + j, N = n0 + nl.
#pragma unroll
  for (int j = 0; j < 8; ++j) {
    int m = m0 + khalf * 8 + j;
    if (m < M_ROWS) P[(size_t)m * N_DIM + n0 + nl] = c[j];
  }
}

// ---------------------------------------------------------------------------
// Phase 2: one wave per edge; both direction scores + per-row atomic max.
//   z_ui = (Q_u[r]·K_i[c] - log(-log(noise_ui))) / tau  (segment = row r)
//   z_iu = (Q_i[c]·K_u[r] - log(-log(noise_iu))) / tau  (segment = col c)
// P row r: [0:128)=Q(r), [128:256)=K(r); items live at rows U_ROWS+c.
// ---------------------------------------------------------------------------
__global__ void edge_score_kernel(const float* __restrict__ P,
                                  const int* __restrict__ rows,
                                  const int* __restrict__ cols,
                                  const float* __restrict__ noise_ui,
                                  const float* __restrict__ noise_iu,
                                  float* __restrict__ z_ui, float* __restrict__ z_iu,
                                  float* __restrict__ rmax_u, float* __restrict__ rmax_i) {
  const int  wave = threadIdx.x >> 5;
  const int  lane = threadIdx.x & 31;
  const long e    = (long)blockIdx.x * (blockDim.x >> 5) + wave;
  if (e >= E_EDGES) return;

  const int r = rows[e];
  const int c = cols[e];
  const float4* pr = (const float4*)(P + (size_t)r * N_DIM);
  const float4* pc = (const float4*)(P + (size_t)(U_ROWS + c) * N_DIM);

  float4 aq = pr[lane];        // Q_u[r], dims 4*lane..+3
  float4 ak = pr[lane + 32];   // K_u[r]
  float4 bq = pc[lane];        // Q_i[c]
  float4 bk = pc[lane + 32];   // K_i[c]

  float wui = aq.x * bk.x + aq.y * bk.y + aq.z * bk.z + aq.w * bk.w;
  float wiu = bq.x * ak.x + bq.y * ak.y + bq.z * ak.z + bq.w * ak.w;
#pragma unroll
  for (int off = 16; off > 0; off >>= 1) {
    wui += __shfl_xor(wui, off, 32);
    wiu += __shfl_xor(wiu, off, 32);
  }

  if (lane == 0) {
    float zu = (wui - __logf(-__logf(noise_ui[e]))) * INV_TAU;
    z_ui[e] = zu;
    atomicMaxFloat(rmax_u + r, zu);
    float zi = (wiu - __logf(-__logf(noise_iu[e]))) * INV_TAU;
    z_iu[e] = zi;
    atomicMaxFloat(rmax_i + c, zi);
  }
}

// ---------------------------------------------------------------------------
// Phase 3: e = exp(z - rowmax); accumulate per-row sums.
// ---------------------------------------------------------------------------
__global__ void edge_exp_kernel(const int* __restrict__ rows, const int* __restrict__ cols,
                                float* __restrict__ z_ui, float* __restrict__ z_iu,
                                const float* __restrict__ rmax_u, const float* __restrict__ rmax_i,
                                float* __restrict__ rsum_u, float* __restrict__ rsum_i) {
  const long e = (long)blockIdx.x * blockDim.x + threadIdx.x;
  if (e >= E_EDGES) return;
  const int r = rows[e];
  const int c = cols[e];
  float eu = __expf(z_ui[e] - rmax_u[r]);
  z_ui[e] = eu;
  atomicAdd(rsum_u + r, eu);
  float ei = __expf(z_iu[e] - rmax_i[c]);
  z_iu[e] = ei;
  atomicAdd(rsum_i + c, ei);
}

// ---------------------------------------------------------------------------
// Phase 4: normalize -> d_out = [vals_ui (E) | vals_iu (E)]
// ---------------------------------------------------------------------------
__global__ void edge_norm_kernel(const int* __restrict__ rows, const int* __restrict__ cols,
                                 const float* __restrict__ z_ui, const float* __restrict__ z_iu,
                                 const float* __restrict__ rsum_u, const float* __restrict__ rsum_i,
                                 float* __restrict__ out) {
  const long e = (long)blockIdx.x * blockDim.x + threadIdx.x;
  if (e >= E_EDGES) return;
  out[e]           = z_ui[e] / rsum_u[rows[e]];
  out[E_EDGES + e] = z_iu[e] / rsum_i[cols[e]];
}

// ---------------------------------------------------------------------------
extern "C" void kernel_launch(void* const* d_in, const int* in_sizes, int n_in,
                              void* d_out, int out_size, void* d_ws, size_t ws_size,
                              hipStream_t stream) {
  const float* user_embed = (const float*)d_in[0];
  const float* item_embed = (const float*)d_in[1];
  const float* Wq         = (const float*)d_in[2];
  const float* bq         = (const float*)d_in[3];
  const float* Wk         = (const float*)d_in[4];
  const float* bk         = (const float*)d_in[5];
  const int*   ui_rows    = (const int*)d_in[6];
  const int*   ui_cols    = (const int*)d_in[7];
  const float* noise_ui   = (const float*)d_in[8];
  const float* noise_iu   = (const float*)d_in[9];
  float*       out        = (float*)d_out;

  // Workspace carve-up (~90.3 MB total).
  float*  P      = (float*)d_ws;                         // 75000*256 f32 = 76.8 MB
  float*  z_ui   = P + (size_t)M_ROWS * N_DIM;           // E f32
  float*  z_iu   = z_ui + E_EDGES;                       // E f32
  float*  rmax_u = z_iu + E_EDGES;                       // 50000
  float*  rsum_u = rmax_u + U_ROWS;                      // 50000
  float*  rmax_i = rsum_u + U_ROWS;                      // 25000
  float*  rsum_i = rmax_i + I_ROWS;                      // 25000
  float*  bc     = rsum_i + I_ROWS;                      // 256
  __bf16* WcT    = (__bf16*)(bc + N_DIM);                // 256*256 bf16 (32B aligned)

  // Phase 0: weights transpose + bias + accumulator init.
  {
    int threads = 256;
    int work = M_ROWS;                                   // covers 65536 and 75000
    if (work < N_DIM * D_DIM) work = N_DIM * D_DIM;
    int blocks = (work + threads - 1) / threads;
    prep_kernel<<<blocks, threads, 0, stream>>>(Wq, bq, Wk, bk, WcT, bc,
                                                rmax_u, rsum_u, rmax_i, rsum_i);
  }

  // Phase 1: WMMA GEMM. tiles = ceil(75000/16)*16 = 75008 waves, 4 waves/block.
  {
    int m_tiles = (M_ROWS + 15) / 16;                    // 4688
    int tiles   = m_tiles * (N_DIM / 16);                // 75008
    int wpb     = 4;
    int blocks  = (tiles + wpb - 1) / wpb;               // 18752
    gemm_qk_kernel<<<blocks, wpb * 32, 0, stream>>>(user_embed, item_embed, WcT, bc, P);
  }

  // Phase 2: edge scores, wave per edge (8 waves / 256-thread block).
  {
    int wpb    = 8;
    long blocks = (E_EDGES + wpb - 1) / wpb;             // 200000
    edge_score_kernel<<<(int)blocks, wpb * 32, 0, stream>>>(
        P, ui_rows, ui_cols, noise_ui, noise_iu, z_ui, z_iu, rmax_u, rmax_i);
  }

  // Phase 3: exp + row sums.
  {
    int threads = 256;
    long blocks = (E_EDGES + threads - 1) / threads;
    edge_exp_kernel<<<(int)blocks, threads, 0, stream>>>(
        ui_rows, ui_cols, z_ui, z_iu, rmax_u, rmax_i, rsum_u, rsum_i);
  }

  // Phase 4: normalize into d_out.
  {
    int threads = 256;
    long blocks = (E_EDGES + threads - 1) / threads;
    edge_norm_kernel<<<(int)blocks, threads, 0, stream>>>(
        ui_rows, ui_cols, z_ui, z_iu, rsum_u, rsum_i, out);
  }
}
